// BiMambaMixerModel_39359080301032
// MI455X (gfx1250) — matrix operand order repaired and checked by reading, not compiled, not run on hardware
//
#include <hip/hip_runtime.h>
#include <hip/hip_bf16.h>
#include <math.h>

// ---------------------------------------------------------------------------
// Bi-directional Mamba (2 layers) for MI455X / gfx1250, wave32 + WMMA bf16.
//   B=2, L=1024, D_MODEL=256, D_INNER=512, D_STATE=16, DT_RANK=16, D_CONV=4
// GEMMs via v_wmma_f32_16x16x32_bf16; each wave computes a 16x(NB*16) strip,
// reusing the A fragment across NB accumulators (NB=4 or 3).
// Scan: 1 thread per (b,d) channel, 16 states in registers, B/C via LDS.
// ---------------------------------------------------------------------------

#define BB      2
#define LL      1024
#define DM      256
#define DI      512
#define NXZ     1024         // 2*DI
#define DS      16
#define DTR     16
#define NDB     48           // DTR + 2*DS
#define MM      (BB*LL)      // 2048 rows

typedef __attribute__((ext_vector_type(16))) __bf16 v16bf;
typedef __attribute__((ext_vector_type(8)))  __bf16 v8bf;
typedef __attribute__((ext_vector_type(8)))  float  v8f;

// ------------------------------ WMMA GEMM ----------------------------------
// D[M,N] (f32) = A[M,K] (bf16, row-major, lda) * W[N,K]^T (bf16, row-major, ldw)
// K multiple of 32; M multiple of 16; N multiple of NB*16.
// One wave -> 16 x (NB*16) output strip; A fragment reused across NB wmmas.
template <int NB>
__global__ void bimamba_gemm_bf16(const __bf16* __restrict__ A, int lda,
                                  const __bf16* __restrict__ W, int ldw,
                                  float* __restrict__ D, int ldd,
                                  int M, int N, int K)
{
    const int lane   = threadIdx.x & 31;
    const int wave   = threadIdx.x >> 5;
    const int tilesN = N >> 4;                  // divisible by NB
    const int grpsN  = tilesN / NB;
    const int group  = blockIdx.x * 8 + wave;
    const int total  = (M >> 4) * grpsN;
    if (group >= total) return;                 // wave-uniform; EXEC stays all-1s

    const int tm   = (group / grpsN) << 4;
    const int tn0  = (group % grpsN) * (NB << 4);
    const int half = lane >> 4;                 // 0: lanes 0-15, 1: lanes 16-31
    const int l16  = lane & 15;

    // A fragment: lane holds row tm+l16; 16 contiguous K starting at 0/16
    const __bf16* arow = A + (size_t)(tm + l16) * lda + (half ? 16 : 0);
    // B fragments: lane holds col tn0+u*16+l16; 16 contiguous K starting at 0/16
    const __bf16* brow[NB];
#pragma unroll
    for (int u = 0; u < NB; ++u)
        brow[u] = W + (size_t)(tn0 + (u << 4) + l16) * ldw + (half ? 16 : 0);

    v8f c[NB];
#pragma unroll
    for (int u = 0; u < NB; ++u) c[u] = (v8f){};

    for (int k = 0; k < K; k += 32) {
        __builtin_prefetch(arow + k + 128, 0, 3);       // global_prefetch_b8
#pragma unroll
        for (int u = 0; u < NB; ++u)
            __builtin_prefetch(brow[u] + k + 128, 0, 3);

        v8bf a0 = *(const v8bf*)(arow + k);
        v8bf a1 = *(const v8bf*)(arow + k + 8);
        v16bf a;
#pragma unroll
        for (int j = 0; j < 8; ++j) { a[j] = a0[j]; a[j + 8] = a1[j]; }

#pragma unroll
        for (int u = 0; u < NB; ++u) {
            v16bf b = *(const v16bf*)(brow[u] + k);
            // (neg_a, A, neg_b, B, c_mod, C, reuse_a, reuse_b)
            c[u] = __builtin_amdgcn_wmma_f32_16x16x32_bf16(false, a, false, b,
                                                           (short)0, c[u], false, false);
        }
    }

    float* drow = D + (size_t)(tm + (half ? 8 : 0)) * ldd + l16;
#pragma unroll
    for (int u = 0; u < NB; ++u)
#pragma unroll
        for (int r = 0; r < 8; ++r)
            drow[(size_t)r * ldd + tn0 + (u << 4)] = c[u][r];
}

// ------------------------------ LayerNorm ----------------------------------
// one block (256 threads) per row; D_MODEL == 256
__global__ void bimamba_layernorm(const float* __restrict__ x,
                                  const float* __restrict__ w,
                                  const float* __restrict__ b,
                                  float* __restrict__ y,
                                  __bf16* __restrict__ yb)
{
    __shared__ float red[DM];
    const int row = blockIdx.x;
    const int tid = threadIdx.x;
    const float v = x[(size_t)row * DM + tid];

    red[tid] = v;           __syncthreads();
    for (int s = 128; s > 0; s >>= 1) { if (tid < s) red[tid] += red[tid + s]; __syncthreads(); }
    const float mu = red[0] * (1.0f / DM);
    __syncthreads();

    const float d = v - mu;
    red[tid] = d * d;       __syncthreads();
    for (int s = 128; s > 0; s >>= 1) { if (tid < s) red[tid] += red[tid + s]; __syncthreads(); }
    const float var = red[0] * (1.0f / DM);

    const float o = d * rsqrtf(var + 1e-5f) * w[tid] + b[tid];
    y[(size_t)row * DM + tid] = o;
    if (yb) yb[(size_t)row * DM + tid] = (__bf16)o;
}

// ---------------------- causal conv (K=4) + SiLU ---------------------------
// reads x = xz[:, :, 0:DI] with optional time reversal; writes dir-local time
__global__ void bimamba_conv_silu(const float* __restrict__ xz,
                                  const float* __restrict__ cw,   // [DI,4]
                                  const float* __restrict__ cb,   // [DI]
                                  float* __restrict__ xc,
                                  __bf16* __restrict__ xcb,
                                  int rev)
{
    const int idx = blockIdx.x * 256 + threadIdx.x;       // < BB*LL*DI
    const int ch  = idx & (DI - 1);
    const int t   = (idx >> 9) & (LL - 1);
    const int bb  = idx >> 19;

    float acc = cb[ch];
#pragma unroll
    for (int i = 0; i < 4; ++i) {
        const int ts = t - 3 + i;
        if (ts >= 0) {
            const int tm = rev ? (LL - 1 - ts) : ts;
            acc += xz[((size_t)bb * LL + tm) * NXZ + ch] * cw[ch * 4 + i];
        }
    }
    const float s = acc / (1.0f + __expf(-acc));          // SiLU
    xc [(size_t)idx] = s;
    xcb[(size_t)idx] = (__bf16)s;
}

// ------------------------- small utility kernels ---------------------------
__global__ void bimamba_f2bf(const float* __restrict__ in, __bf16* __restrict__ out, int n)
{
    const int i = blockIdx.x * 256 + threadIdx.x;
    if (i < n) out[i] = (__bf16)in[i];
}

__global__ void bimamba_zero_f32(float* __restrict__ p, int n)
{
    const int i = blockIdx.x * 256 + threadIdx.x;
    if (i < n) p[i] = 0.0f;
}

// pad dt_w [rows,16] -> bf16 [rows,32] (cols 16..31 = 0)
__global__ void bimamba_dtw_pad(const float* __restrict__ in, __bf16* __restrict__ out, int rows)
{
    const int i = blockIdx.x * 256 + threadIdx.x;
    if (i >= rows * 32) return;
    const int r = i >> 5, j = i & 31;
    out[i] = (j < 16) ? (__bf16)in[r * 16 + j] : (__bf16)0.0f;
}

// dt input: xdbl[:, 0:16] -> bf16 [M,32] zero-padded
__global__ void bimamba_dtprep(const float* __restrict__ xdbl, __bf16* __restrict__ dtin)
{
    const int i = blockIdx.x * 256 + threadIdx.x;
    if (i >= MM * 32) return;
    const int m = i >> 5, j = i & 31;
    dtin[i] = (j < 16) ? (__bf16)xdbl[(size_t)m * NDB + j] : (__bf16)0.0f;
}

// --------------------------- selective scan --------------------------------
// 64 threads/block, 1 thread per channel; 16 WGP-spread blocks total.
// gsum[b, t_orig, ch] += (y + x*Dp) * silu(z)   (dir-aware write index)
__global__ void bimamba_scan(const float* __restrict__ dtraw,   // [M,DI] dir-local
                             const float* __restrict__ dtb,     // [DI]
                             const float* __restrict__ Alog,    // [DI,DS]
                             const float* __restrict__ Dp,      // [DI]
                             const float* __restrict__ xc,      // [M,DI] dir-local
                             const float* __restrict__ xdbl,    // [M,NDB] dir-local
                             const float* __restrict__ xz,      // [M,NXZ] original time
                             float* __restrict__ gsum,          // [M,DI] original time
                             int rev)
{
    __shared__ float bc[2 * DS];                                 // Bm | C per step
    const int bb  = blockIdx.x >> 3;
    const int ch  = ((blockIdx.x & 7) << 6) + threadIdx.x;

    float Av[DS], st[DS];
#pragma unroll
    for (int n = 0; n < DS; ++n) {
        Av[n] = -__expf(Alog[(size_t)ch * DS + n]);
        st[n] = 0.0f;
    }
    const float dtbv = dtb[ch];
    const float dpv  = Dp[ch];

    for (int t = 0; t < LL; ++t) {
        if (threadIdx.x < 2 * DS)
            bc[threadIdx.x] = xdbl[((size_t)bb * LL + t) * NDB + DTR + threadIdx.x];
        __syncthreads();

        float dtv = dtraw[((size_t)bb * LL + t) * DI + ch] + dtbv;
        dtv = (dtv > 20.0f) ? dtv : log1pf(__expf(dtv));         // softplus
        const float xv = xc[((size_t)bb * LL + t) * DI + ch];
        const float du = dtv * xv;

        float y = 0.0f;
#pragma unroll
        for (int n = 0; n < DS; ++n) {
            const float dA = __expf(dtv * Av[n]);
            st[n] = dA * st[n] + du * bc[n];
            y    += st[n] * bc[DS + n];
        }
        y += xv * dpv;

        const int to = rev ? (LL - 1 - t) : t;
        const float zv = xz[((size_t)bb * LL + to) * NXZ + DI + ch];
        gsum[((size_t)bb * LL + to) * DI + ch] += y * (zv / (1.0f + __expf(-zv)));
        __syncthreads();
    }
}

// ------------------------------- host side ---------------------------------
extern "C" void kernel_launch(void* const* d_in, const int* in_sizes, int n_in,
                              void* d_out, int out_size, void* d_ws, size_t ws_size,
                              hipStream_t stream)
{
    (void)in_sizes; (void)n_in; (void)out_size; (void)ws_size;

    const float* inp      = (const float*)d_in[0];
    const float* ln_w     = (const float*)d_in[1];
    const float* ln_b     = (const float*)d_in[2];
    const float* in_w     = (const float*)d_in[3];
    const float* out_w    = (const float*)d_in[4];
    const float* conv_w_f = (const float*)d_in[5];
    const float* conv_b_f = (const float*)d_in[6];
    const float* xproj_f  = (const float*)d_in[7];
    const float* dt_w_f   = (const float*)d_in[8];
    const float* dt_b_f   = (const float*)d_in[9];
    const float* Alog_f   = (const float*)d_in[10];
    const float* Dp_f     = (const float*)d_in[11];
    const float* conv_w_r = (const float*)d_in[12];
    const float* conv_b_r = (const float*)d_in[13];
    const float* xproj_r  = (const float*)d_in[14];
    const float* dt_w_r   = (const float*)d_in[15];
    const float* dt_b_r   = (const float*)d_in[16];
    const float* Alog_r   = (const float*)d_in[17];
    const float* Dp_r     = (const float*)d_in[18];
    const float* normf_w  = (const float*)d_in[19];
    const float* normf_b  = (const float*)d_in[20];

    // ---- workspace arena (256B aligned slices) ----
    char* base = (char*)d_ws;
    size_t off = 0;
    auto alloc = [&](size_t bytes) -> void* {
        void* p = base + off;
        off = (off + bytes + 255) & ~(size_t)255;
        return p;
    };
    float*  hn    = (float*) alloc((size_t)MM * DM  * 4);
    __bf16* hnb   = (__bf16*)alloc((size_t)MM * DM  * 2);
    float*  xz    = (float*) alloc((size_t)MM * NXZ * 4);
    float*  xc    = (float*) alloc((size_t)MM * DI  * 4);
    __bf16* xcb   = (__bf16*)alloc((size_t)MM * DI  * 2);
    float*  xdbl  = (float*) alloc((size_t)MM * NDB * 4);
    __bf16* dtin  = (__bf16*)alloc((size_t)MM * 32  * 2);
    float*  dtraw = (float*) alloc((size_t)MM * DI  * 4);
    float*  gsum  = (float*) alloc((size_t)MM * DI  * 4);
    __bf16* gsb   = (__bf16*)alloc((size_t)MM * DI  * 2);
    float*  hbuf  = (float*) alloc((size_t)MM * DM  * 4);
    __bf16* inwb  = (__bf16*)alloc((size_t)2 * NXZ * DM * 2);
    __bf16* outwb = (__bf16*)alloc((size_t)2 * DM  * DI * 2);
    __bf16* xpfb  = (__bf16*)alloc((size_t)2 * NDB * DI * 2);
    __bf16* xprb  = (__bf16*)alloc((size_t)2 * NDB * DI * 2);
    __bf16* dtwfb = (__bf16*)alloc((size_t)2 * DI * 32 * 2);
    __bf16* dtwrb = (__bf16*)alloc((size_t)2 * DI * 32 * 2);

    // ---- one-time weight conversions (deterministic, on stream) ----
    auto cvt = [&](const float* s, __bf16* d, int n) {
        bimamba_f2bf<<<(n + 255) / 256, 256, 0, stream>>>(s, d, n);
    };
    cvt(in_w,    inwb,  2 * NXZ * DM);
    cvt(out_w,   outwb, 2 * DM  * DI);
    cvt(xproj_f, xpfb,  2 * NDB * DI);
    cvt(xproj_r, xprb,  2 * NDB * DI);
    bimamba_dtw_pad<<<(2 * DI * 32 + 255) / 256, 256, 0, stream>>>(dt_w_f, dtwfb, 2 * DI);
    bimamba_dtw_pad<<<(2 * DI * 32 + 255) / 256, 256, 0, stream>>>(dt_w_r, dtwrb, 2 * DI);

    const float* hcur = inp;
    for (int i = 0; i < 2; ++i) {
        // LayerNorm -> hn (f32) + hnb (bf16)
        bimamba_layernorm<<<MM, DM, 0, stream>>>(hcur, ln_w + i * DM, ln_b + i * DM, hn, hnb);

        // in-proj: [2048,256] x [1024,256]^T -> xz [2048,1024]   (NB=4 strips)
        bimamba_gemm_bf16<4><<<(MM / 16) * (NXZ / 64) / 8, 256, 0, stream>>>(
            hnb, DM, inwb + (size_t)i * NXZ * DM, DM, xz, NXZ, MM, NXZ, DM);

        bimamba_zero_f32<<<(MM * DI + 255) / 256, 256, 0, stream>>>(gsum, MM * DI);

        for (int dir = 0; dir < 2; ++dir) {
            const float*  cw  = dir ? conv_w_r : conv_w_f;
            const float*  cb  = dir ? conv_b_r : conv_b_f;
            const __bf16* xpw = dir ? xprb     : xpfb;
            const __bf16* dtw = dir ? dtwrb    : dtwfb;
            const float*  dtb = dir ? dt_b_r   : dt_b_f;
            const float*  Alg = dir ? Alog_r   : Alog_f;
            const float*  dp  = dir ? Dp_r     : Dp_f;

            bimamba_conv_silu<<<(MM * DI) / 256, 256, 0, stream>>>(
                xz, cw + (size_t)i * DI * 4, cb + (size_t)i * DI, xc, xcb, dir);

            // x-proj: [2048,512] x [48,512]^T -> xdbl [2048,48]  (NB=3: full strip)
            bimamba_gemm_bf16<3><<<(MM / 16) / 8, 256, 0, stream>>>(
                xcb, DI, xpw + (size_t)i * NDB * DI, DI, xdbl, NDB, MM, NDB, DI);

            bimamba_dtprep<<<(MM * 32) / 256, 256, 0, stream>>>(xdbl, dtin);

            // dt-proj (K padded 16->32): [2048,32] x [512,32]^T -> dtraw [2048,512]
            bimamba_gemm_bf16<4><<<(MM / 16) * (DI / 64) / 8, 256, 0, stream>>>(
                dtin, 32, dtw + (size_t)i * DI * 32, 32, dtraw, DI, MM, DI, 32);

            bimamba_scan<<<BB * 8, 64, 0, stream>>>(
                dtraw, dtb + (size_t)i * DI, Alg + (size_t)i * DI * DS,
                dp + (size_t)i * DI, xc, xdbl, xz, gsum, dir);
        }

        // out-proj on summed gated output: [2048,512] x [256,512]^T -> hbuf
        bimamba_f2bf<<<(MM * DI + 255) / 256, 256, 0, stream>>>(gsum, gsb, MM * DI);
        bimamba_gemm_bf16<4><<<(MM / 16) * (DM / 64) / 8, 256, 0, stream>>>(
            gsb, DI, outwb + (size_t)i * DM * DI, DI, hbuf, DM, MM, DM, DI);

        hcur = hbuf;
    }

    // final LayerNorm -> d_out (f32)
    bimamba_layernorm<<<MM, DM, 0, stream>>>(hbuf, normf_w, normf_b, (float*)d_out, nullptr);
}
